// update_e_5952824672726
// MI455X (gfx1250) — matrix operand order repaired and checked by reading, not compiled, not run on hardware
//
#include <hip/hip_runtime.h>
#include <hip/hip_bf16.h>

typedef __attribute__((ext_vector_type(16))) _Float16 v16h;
typedef __attribute__((ext_vector_type(8)))  _Float16 v8h;
typedef __attribute__((ext_vector_type(4)))  _Float16 v4h;
typedef __attribute__((ext_vector_type(8)))  float    v8f;
typedef __attribute__((ext_vector_type(4)))  float    v4f;

__device__ __forceinline__ float selu_f(float x) {
  const float alpha = 1.6732632423543772848170429916717f;
  const float scale = 1.0507009873554804934193349852946f;
  return scale * (x > 0.f ? x : alpha * (__expf(x) - 1.f));
}

// ---------------------------------------------------------------- elementwise
// n must be a multiple of 4 (E*H and E*INT are).
__global__ void cvt_f2h4_kernel(const float* __restrict__ in, _Float16* __restrict__ out, int n4) {
  int i = blockIdx.x * blockDim.x + threadIdx.x;
  if (i < n4) {
    v4f x = *(const v4f*)(in + (size_t)i * 4);
    v4h y;
    y[0] = (_Float16)x[0]; y[1] = (_Float16)x[1];
    y[2] = (_Float16)x[2]; y[3] = (_Float16)x[3];
    *(v4h*)(out + (size_t)i * 4) = y;
  }
}

__global__ void zero_f4_kernel(float* __restrict__ p, int n4) {
  int i = blockIdx.x * blockDim.x + threadIdx.x;
  if (i < n4) {
    v4f z = {};
    *(v4f*)(p + (size_t)i * 4) = z;
  }
}

__global__ void mul4_kernel(const float* __restrict__ a, const float* __restrict__ b,
                            float* __restrict__ out, int n4) {
  int i = blockIdx.x * blockDim.x + threadIdx.x;
  if (i < n4) {
    v4f x = *(const v4f*)(a + (size_t)i * 4);
    v4f y = *(const v4f*)(b + (size_t)i * 4);
    v4f z = { x[0]*y[0], x[1]*y[1], x[2]*y[2], x[3]*y[3] };
    *(v4f*)(out + (size_t)i * 4) = z;
  }
}

// -------------------------------------------------- weight -> B-fragment pack
// Repack f32 row-major W[K,N] into WMMA B-fragment order (f16):
// linear index = ((kstep*(N/16) + ntile)*32 + lane)*16 + i
// half i of lane holds B[k = kstep*32 + i + 16*(lane>=16)][n = ntile*16 + (lane&15)]
__global__ void pack_b_kernel(const float* __restrict__ W, _Float16* __restrict__ out,
                              int K, int N) {
  int idx = blockIdx.x * blockDim.x + threadIdx.x;
  if (idx >= K * N) return;
  int i    = idx & 15;
  int lane = (idx >> 4) & 31;
  int rest = idx >> 9;
  int ntiles = N >> 4;
  int nt    = rest % ntiles;
  int kstep = rest / ntiles;
  int k = (kstep << 5) + i + ((lane >> 4) << 4);
  int n = (nt << 4) + (lane & 15);
  out[idx] = (_Float16)W[k * N + n];
}

// ------------------------------------------------------------- WMMA GEMM
// Two 16x16 output tiles (stacked in M) per wave -> B fragment reused twice,
// 2 v_wmma per k-step. A: row-major f16 [M,K] (M multiple of 32). Bf: packed.
// Straight-line templated epilogue: (+bias) -> selu -> (*mul) -> (+resid).
template <int K, bool BIAS, bool SELU, bool MUL, bool RES, bool OF, bool OH>
__global__ void gemm_wmma_kernel(const _Float16* __restrict__ A,
                                 const _Float16* __restrict__ Bf,
                                 const float* __restrict__ bias,
                                 const float* __restrict__ mul,
                                 const float* __restrict__ resid,
                                 float* __restrict__ outF,
                                 _Float16* __restrict__ outH,
                                 int M, int N) {
  const int lane = threadIdx.x & 31;
  const int wave = threadIdx.x >> 5;
  const int ntiles = N >> 4;
  const int tiles = (M >> 5) * ntiles;          // M-tile pairs * N-tiles
  const int tile = blockIdx.x * (blockDim.x >> 5) + wave;
  if (tile >= tiles) return;                    // wave-uniform: EXEC stays all-ones
  const int mp = tile / ntiles;
  const int nt = tile - mp * ntiles;
  const int lh = lane >> 4;                     // 0 or 1 (half-wave)
  const int ll = lane & 15;
  const int m0 = (mp << 5) + ll;
  const _Float16* arow0 = A + (size_t)m0 * K;
  const _Float16* arow1 = A + (size_t)(m0 + 16) * K;
  const _Float16* bbase = Bf + ((size_t)nt << 9) + ((size_t)lane << 4);
  v8f acc0 = {}, acc1 = {};
#pragma unroll
  for (int ks = 0; ks < K; ks += 32) {
    // B 32x16 fragment: one contiguous 32B load per lane
    v16h bf = *(const v16h*)(bbase + ((size_t)(ks >> 5) * ntiles << 9));
    // A 16x32 fragments: lanes 0-15 hold K 0..15, lanes 16-31 hold 16..31
    v8h a00 = *(const v8h*)(arow0 + ks + lh * 16);
    v8h a01 = *(const v8h*)(arow0 + ks + 8 + lh * 16);
    v8h a10 = *(const v8h*)(arow1 + ks + lh * 16);
    v8h a11 = *(const v8h*)(arow1 + ks + 8 + lh * 16);
    v16h af0, af1;
#pragma unroll
    for (int i = 0; i < 8; ++i) {
      af0[i] = a00[i]; af0[i + 8] = a01[i];
      af1[i] = a10[i]; af1[i + 8] = a11[i];
    }
    acc0 = __builtin_amdgcn_wmma_f32_16x16x32_f16(false, af0, false, bf,
                                                  (short)0, acc0, false, false);
    acc1 = __builtin_amdgcn_wmma_f32_16x16x32_f16(false, af1, false, bf,
                                                  (short)0, acc1, false, false);
  }
  const int col = (nt << 4) + ll;
  const float bv = BIAS ? bias[col] : 0.f;
#pragma unroll
  for (int half = 0; half < 2; ++half) {
    const v8f acc = half ? acc1 : acc0;
    const int rbase = (mp << 5) + (half << 4) + lh * 8;  // C/D layout: vgpr r -> M=r (+8 hi lanes)
#pragma unroll
    for (int r = 0; r < 8; ++r) {
      const size_t o = (size_t)(rbase + r) * N + col;
      float v = acc[r];
      if (BIAS) v += bv;
      if (SELU) v  = selu_f(v);
      if (MUL)  v *= mul[o];
      if (RES)  v += resid[o];
      if (OF) outF[o] = v;
      if (OH) outH[o] = (_Float16)v;
    }
  }
}

template <int K, bool BIAS, bool SELU, bool MUL, bool RES, bool OF, bool OH>
static inline void launch_gemm(const _Float16* A, const _Float16* Bf, const float* bias,
                               const float* mul, const float* resid, float* outF, _Float16* outH,
                               int M, int N, hipStream_t s) {
  int tiles = (M / 32) * (N / 16);
  int grid = (tiles + 7) / 8;
  gemm_wmma_kernel<K, BIAS, SELU, MUL, RES, OF, OH>
      <<<grid, 256, 0, s>>>(A, Bf, bias, mul, resid, outF, outH, M, N);
}

// ----------------------------------------------------------------- rbf kernel
// rbf  = (rbf0 @ Wr1[6,8]) @ Wr2[8,128]   -> [E,128]
// rbf2 =  rbf0 @ Wr [6,128]               -> [E,128]
__global__ void rbf_kernel(const float* __restrict__ rbf0,
                           const float* __restrict__ Wr1, const float* __restrict__ Wr2,
                           const float* __restrict__ Wr,
                           float* __restrict__ rbf, float* __restrict__ rbf2) {
  const int e = blockIdx.x;
  const int h = threadIdx.x;       // blockDim = 128
  __shared__ float s_in[8];
  __shared__ float s_t[8];
  if (h < 6) s_in[h] = rbf0[(size_t)e * 6 + h];
  __syncthreads();
  if (h < 8) {
    float a = 0.f;
#pragma unroll
    for (int k = 0; k < 6; ++k) a += s_in[k] * Wr1[k * 8 + h];
    s_t[h] = a;
  }
  __syncthreads();
  float v = 0.f;
#pragma unroll
  for (int j = 0; j < 8; ++j) v += s_t[j] * Wr2[j * 128 + h];
  rbf[(size_t)e * 128 + h] = v;
  float v2 = 0.f;
#pragma unroll
  for (int k = 0; k < 6; ++k) v2 += s_in[k] * Wr[k * 128 + h];
  rbf2[(size_t)e * 128 + h] = v2;
}

// --------------------------------------------------------------- triplet pass
// For each triplet t: sbf_p = (sbf_row@Ws1)@Ws2, t_p = (t_row@Wt1)@Wt2,
// val = x_kj[idx_kj[t]] * sbf_p * t_p, atomic segment-sum into accum[idx_ji[t]].
// 4 triplets per 256-thread block; basis weights staged in LDS; next t-row
// prefetched (global_prefetch_b8) since the t stream (588 MB) is the roofline.
__global__ void triplet_kernel(const float* __restrict__ sbf, const float* __restrict__ tin,
                               const float* __restrict__ Ws1, const float* __restrict__ Ws2,
                               const float* __restrict__ Wt1, const float* __restrict__ Wt2,
                               const float* __restrict__ xkj,
                               const int* __restrict__ idx_kj, const int* __restrict__ idx_ji,
                               float* __restrict__ accum, int T) {
  __shared__ float sWt1[294 * 8];
  __shared__ float sWs1[42 * 8];
  __shared__ float sWt2[8 * 64];
  __shared__ float sWs2[8 * 64];
  __shared__ float s_t[4][294];
  __shared__ float s_s[4][48];
  __shared__ float s_red[4][64];
  __shared__ float s_tt[4][8];
  __shared__ float s_ts[4][8];
  const int tid = threadIdx.x;
  for (int i = tid; i < 294 * 8; i += 256) sWt1[i] = Wt1[i];
  for (int i = tid; i < 42 * 8;  i += 256) sWs1[i] = Ws1[i];
  for (int i = tid; i < 512;     i += 256) { sWt2[i] = Wt2[i]; sWs2[i] = Ws2[i]; }
  const int g = tid >> 6, h = tid & 63;
  const int tr = blockIdx.x * 4 + g;
  const bool valid = tr < T;
  if (valid) {
    const float* trow = tin + (size_t)tr * 294;
    for (int k = h; k < 294; k += 64) s_t[g][k] = trow[k];
    if (h < 42) s_s[g][h] = sbf[(size_t)tr * 42 + h];
    if (tr + 4 < T) {
      // 64 lanes spread across the next row's 1176 bytes -> covers all its lines
      __builtin_prefetch(tin + (size_t)(tr + 4) * 294 + ((h * 294) >> 6), 0, 1);
    }
  }
  __syncthreads();
  const int j = h >> 3, sub = h & 7;
  float at = 0.f;
  if (valid) for (int k = sub; k < 294; k += 8) at += s_t[g][k] * sWt1[k * 8 + j];
  s_red[g][h] = at;
  __syncthreads();
  if (h < 8) { float s = 0.f; for (int q = 0; q < 8; ++q) s += s_red[g][h * 8 + q]; s_tt[g][h] = s; }
  __syncthreads();
  float as = 0.f;
  if (valid) for (int k = sub; k < 42; k += 8) as += s_s[g][k] * sWs1[k * 8 + j];
  s_red[g][h] = as;
  __syncthreads();
  if (h < 8) { float s = 0.f; for (int q = 0; q < 8; ++q) s += s_red[g][h * 8 + q]; s_ts[g][h] = s; }
  __syncthreads();
  if (valid) {
    float sp = 0.f, tp = 0.f;
#pragma unroll
    for (int q = 0; q < 8; ++q) {
      sp += s_ts[g][q] * sWs2[q * 64 + h];
      tp += s_tt[g][q] * sWt2[q * 64 + h];
    }
    const int ekj = idx_kj[tr];
    const int eji = idx_ji[tr];
    const float val = xkj[(size_t)ekj * 64 + h] * sp * tp;
    atomicAdd(accum + (size_t)eji * 64 + h, val);
  }
}

// ------------------------------------------------------------------- launcher
extern "C" void kernel_launch(void* const* d_in, const int* in_sizes, int n_in,
                              void* d_out, int out_size, void* d_ws, size_t ws_size,
                              hipStream_t stream) {
  const float* x1      = (const float*)d_in[0];
  const float* rbf0    = (const float*)d_in[1];
  const float* sbf     = (const float*)d_in[2];
  const float* tin     = (const float*)d_in[3];
  /* d_in[4] = q_r : unused by reference */
  const float* W_rbf1  = (const float*)d_in[5];
  const float* W_rbf2  = (const float*)d_in[6];
  const float* W_sbf1  = (const float*)d_in[7];
  const float* W_sbf2  = (const float*)d_in[8];
  const float* W_t1    = (const float*)d_in[9];
  const float* W_t2    = (const float*)d_in[10];
  const float* W_rbf   = (const float*)d_in[11];
  const float* W_kj    = (const float*)d_in[12];
  const float* b_kj    = (const float*)d_in[13];
  const float* W_ji    = (const float*)d_in[14];
  const float* b_ji    = (const float*)d_in[15];
  const float* W_down  = (const float*)d_in[16];
  const float* W_up    = (const float*)d_in[17];
  const float* W_lin   = (const float*)d_in[18];
  const float* b_lin   = (const float*)d_in[19];
  const float* resb_W1 = (const float*)d_in[20];
  const float* resb_b1 = (const float*)d_in[21];
  const float* resb_W2 = (const float*)d_in[22];
  const float* resb_b2 = (const float*)d_in[23];
  const float* resa_W1 = (const float*)d_in[24];
  const float* resa_b1 = (const float*)d_in[25];
  const float* resa_W2 = (const float*)d_in[26];
  const float* resa_b2 = (const float*)d_in[27];
  const int* idx_kj    = (const int*)d_in[28];
  const int* idx_ji    = (const int*)d_in[29];

  const int H = 128, INTC = 64;
  const int E = in_sizes[0] / H;   // 100000 (multiple of 32)
  const int T = in_sizes[28];      // 500000

  // ------------------------------- workspace carve-up (256B aligned regions)
  char* ws = (char*)d_ws;
  size_t off = 0;
  auto take = [&](size_t bytes) -> char* {
    char* p = ws + off;
    off += (bytes + 255) & ~(size_t)255;
    return p;
  };
  _Float16* H1   = (_Float16*)take((size_t)E * H * 2);    // x1_f16, later e1_f16 ping
  _Float16* H2   = (_Float16*)take((size_t)E * H * 2);    // x_kj_f16, later e1_f16 pong
  _Float16* ACCH = (_Float16*)take((size_t)E * INTC * 2); // segment-sum result f16
  float* F1   = (float*)take((size_t)E * H * 4);          // e1 f32 ping
  float* F2   = (float*)take((size_t)E * H * 4);          // rbf, later e1 f32 pong
  float* XJI  = (float*)take((size_t)E * H * 4);          // x_ji f32
  float* RBF2 = (float*)take((size_t)E * H * 4);          // rbf0 @ W_rbf (for e2)
  float* XKJI = (float*)take((size_t)E * INTC * 4);       // x_kj after down-proj
  float* ACC  = (float*)take((size_t)E * INTC * 4);       // segment-sum accumulator
  _Float16* pWji   = (_Float16*)take(128 * 128 * 2);
  _Float16* pWkj   = (_Float16*)take(128 * 128 * 2);
  _Float16* pWdown = (_Float16*)take(128 * 64 * 2);
  _Float16* pWup   = (_Float16*)take(64 * 128 * 2);
  _Float16* pWlin  = (_Float16*)take(128 * 128 * 2);
  _Float16* pRbW1  = (_Float16*)take(128 * 128 * 2);
  _Float16* pRbW2  = (_Float16*)take(128 * 128 * 2);
  _Float16* pRaW1a = (_Float16*)take(128 * 128 * 2);
  _Float16* pRaW2a = (_Float16*)take(128 * 128 * 2);
  _Float16* pRaW1b = (_Float16*)take(128 * 128 * 2);
  _Float16* pRaW2b = (_Float16*)take(128 * 128 * 2);

  auto pack = [&](const float* W, _Float16* dst, int K, int N) {
    int n = K * N;
    pack_b_kernel<<<(n + 255) / 256, 256, 0, stream>>>(W, dst, K, N);
  };

  // --------------------------------------------------------- prep
  {
    int n4 = (E * H) / 4;
    cvt_f2h4_kernel<<<(n4 + 255) / 256, 256, 0, stream>>>(x1, H1, n4);
  }
  pack(W_ji,   pWji,   128, 128);
  pack(W_kj,   pWkj,   128, 128);
  pack(W_down, pWdown, 128, 64);
  pack(W_up,   pWup,   64, 128);
  pack(W_lin,  pWlin,  128, 128);
  pack(resb_W1,             pRbW1,  128, 128);
  pack(resb_W2,             pRbW2,  128, 128);
  pack(resa_W1,             pRaW1a, 128, 128);
  pack(resa_W2,             pRaW2a, 128, 128);
  pack(resa_W1 + 128 * 128, pRaW1b, 128, 128);
  pack(resa_W2 + 128 * 128, pRaW2b, 128, 128);

  rbf_kernel<<<E, 128, 0, stream>>>(rbf0, W_rbf1, W_rbf2, W_rbf, F2 /*rbf*/, RBF2);

  // --------------------------------------------------------- edge-side GEMMs
  // x_ji = selu(x1 @ W_ji + b_ji)
  launch_gemm<128, true,  true, false, false, true,  false>(H1, pWji, b_ji, nullptr, nullptr, XJI, nullptr, E, H, stream);
  // x_kj = selu(x1 @ W_kj + b_kj) * rbf   (f16 for down-proj)
  launch_gemm<128, true,  true, true,  false, false, true >(H1, pWkj, b_kj, F2, nullptr, nullptr, H2, E, H, stream);
  // x_kj = selu(x_kj @ W_down)  [E,64]
  launch_gemm<128, false, true, false, false, true,  false>(H2, pWdown, nullptr, nullptr, nullptr, XKJI, nullptr, E, INTC, stream);

  // --------------------------------------------------------- triplet pass
  {
    int n4 = (E * INTC) / 4;
    zero_f4_kernel<<<(n4 + 255) / 256, 256, 0, stream>>>(ACC, n4);
  }
  triplet_kernel<<<(T + 3) / 4, 256, 0, stream>>>(sbf, tin, W_sbf1, W_sbf2, W_t1, W_t2,
                                                  XKJI, idx_kj, idx_ji, ACC, T);
  {
    int n4 = (E * INTC) / 4;
    cvt_f2h4_kernel<<<(n4 + 255) / 256, 256, 0, stream>>>(ACC, ACCH, n4);
  }

  // --------------------------------------------------------- up-proj + MLP chain
  // e1 = x_ji + selu(acc @ W_up)
  launch_gemm<64,  false, true, false, true,  true,  true >(ACCH, pWup, nullptr, nullptr, XJI, F1, H1, E, H, stream);
  // res block (before skip): g1 = selu(e1@W1+b1); e1 = e1 + selu(g1@W2+b2)
  launch_gemm<128, true,  true, false, false, false, true >(H1, pRbW1, resb_b1, nullptr, nullptr, nullptr, H2, E, H, stream);
  launch_gemm<128, true,  true, false, true,  true,  true >(H2, pRbW2, resb_b2, nullptr, F1, F2, H1, E, H, stream);
  // e1 = selu(e1 @ W_lin + b_lin) + x1
  launch_gemm<128, true,  true, false, true,  true,  true >(H1, pWlin, b_lin, nullptr, x1, F1, H2, E, H, stream);
  // res block after #0
  launch_gemm<128, true,  true, false, false, false, true >(H2, pRaW1a, resa_b1, nullptr, nullptr, nullptr, H1, E, H, stream);
  launch_gemm<128, true,  true, false, true,  true,  true >(H1, pRaW2a, resa_b2, nullptr, F1, F2, H2, E, H, stream);
  // res block after #1 -> final e1 straight into d_out
  float* out_e1 = (float*)d_out;
  float* out_e2 = (float*)d_out + (size_t)E * H;
  launch_gemm<128, true,  true, false, false, false, true >(H2, pRaW1b, resa_b1 + 128, nullptr, nullptr, nullptr, H1, E, H, stream);
  launch_gemm<128, true,  true, false, true,  true,  false>(H1, pRaW2b, resa_b2 + 128, nullptr, F2, out_e1, nullptr, E, H, stream);

  // e2 = (rbf0 @ W_rbf) * e1
  {
    int n4 = (E * H) / 4;
    mul4_kernel<<<(n4 + 255) / 256, 256, 0, stream>>>(RBF2, out_e1, out_e2, n4);
  }
}
